// VectorQuantizer_84748294685320
// MI455X (gfx1250) — matrix operand order repaired and checked by reading, not compiled, not run on hardware
//
#include <hip/hip_runtime.h>
#include <hip/hip_bf16.h>
#include <stdint.h>

typedef __attribute__((ext_vector_type(16))) _Float16 v16h;
typedef __attribute__((ext_vector_type(8)))  _Float16 v8h;
typedef __attribute__((ext_vector_type(8)))  float    v8f;

#define B_DIM 32
#define D_DIM 128
#define T_DIM 2048
#define K_DIM 4096
#define N_DIM (B_DIM * T_DIM)           // 65536
#define ROWS_PER_BLOCK 64               // 4 waves x 16 rows
#define THREADS 128
#define CODES_PER_CHUNK 64
#define NCHUNK (K_DIM / CODES_PER_CHUNK)
#define ROW_STRIDE 136                  // halves; 272 B rows (16B aligned, conflict-free)
#define ROW_STRIDE_DW (ROW_STRIDE / 2)  // 68 dwords

// ---- CDNA5 async global->LDS path (guarded; falls back to b128 sync copy) ----
#if defined(__has_builtin)
#if __has_builtin(__builtin_amdgcn_global_load_async_to_lds_b128)
#define HAVE_ASYNC_LDS 1
#endif
#endif

#if defined(HAVE_ASYNC_LDS)
// Diagnostic-confirmed pointee type: 'int __attribute__((__vector_size__(4*sizeof(int))))'
typedef int v4i_t __attribute__((vector_size(16)));
typedef __attribute__((address_space(1))) v4i_t* g_v4i_p;   // global (prints as __device__)
typedef __attribute__((address_space(3))) v4i_t* l_v4i_p;   // LDS    (prints as __shared__)
__device__ __forceinline__ void async_copy16(const unsigned* gp, unsigned* lp) {
  // AS casts via integers: generic LDS pointer's low 32 bits are the LDS offset.
  __builtin_amdgcn_global_load_async_to_lds_b128(
      (g_v4i_p)(uintptr_t)gp, (l_v4i_p)(unsigned)(uintptr_t)lp, 0, 0);
}
__device__ __forceinline__ void wait_async0() {
#if __has_builtin(__builtin_amdgcn_s_wait_asynccnt)
  __builtin_amdgcn_s_wait_asynccnt(0);
#else
  asm volatile("s_wait_asynccnt 0x0" ::: "memory");
#endif
}
#else
__device__ __forceinline__ void async_copy16(const unsigned* gp, unsigned* lp) {
  *(uint4*)lp = *(const uint4*)gp;      // b128 load + b128 ds_store fallback
}
__device__ __forceinline__ void wait_async0() {}
#endif

// ---------------------------------------------------------------------------
// Kernel 1: L2-normalize codebook rows to f16; zero loss accumulator.
// ---------------------------------------------------------------------------
__global__ __launch_bounds__(128) void k_norm_codebook(
    const float* __restrict__ W, _Float16* __restrict__ wn, float* __restrict__ loss_accum) {
  __shared__ float sh[128];
  const int k = blockIdx.x;
  const int d = threadIdx.x;
  float v = W[(size_t)k * D_DIM + d];
  sh[d] = v * v;
  __syncthreads();
  #pragma unroll
  for (int s = 64; s > 0; s >>= 1) {
    if (d < s) sh[d] += sh[d + s];
    __syncthreads();
  }
  float scale = 1.0f / fmaxf(sqrtf(sh[0]), 1e-12f);
  wn[(size_t)k * D_DIM + d] = (_Float16)(v * scale);
  if (k == 0 && d == 0) loss_accum[0] = 0.0f;
}

// ---------------------------------------------------------------------------
// Kernel 2: WMMA argmax with double-buffered async codebook staging.
// ---------------------------------------------------------------------------
__global__ __launch_bounds__(THREADS) void k_vq_argmax(
    const float* __restrict__ x, const _Float16* __restrict__ wn, int* __restrict__ out_idx) {
  __shared__ _Float16 xh[ROWS_PER_BLOCK * ROW_STRIDE];        // 17,408 B
  __shared__ _Float16 wh[2 * CODES_PER_CHUNK * ROW_STRIDE];   // 34,816 B
  __shared__ float rowpart[THREADS];

  const int tid = threadIdx.x;
  const int n0  = blockIdx.x * ROWS_PER_BLOCK;
  const int b   = n0 / T_DIM;
  const int t0  = n0 % T_DIM;           // 64 | 2048 -> never straddles a batch
  const float* xbase = x + (size_t)b * D_DIM * T_DIM + t0;

  const unsigned* wn_dw = (const unsigned*)wn;
  unsigned*       wh_dw = (unsigned*)wh;

  // stage one 64-code chunk (64 rows x 64 dwords) as 1024 x 16B segments
  auto stage = [&](int chunk, int buf) {
    unsigned* lbase = wh_dw + buf * (CODES_PER_CHUNK * ROW_STRIDE_DW);
    #pragma unroll
    for (int i = 0; i < 8; ++i) {
      int seg = i * THREADS + tid;      // 0..1023
      int row = seg >> 4;
      int c4  = (seg & 15) * 4;
      async_copy16(wn_dw + ((size_t)(chunk * CODES_PER_CHUNK + row) * (D_DIM / 2) + c4),
                   lbase + row * ROW_STRIDE_DW + c4);
    }
  };

  stage(0, 0);                          // overlap first chunk with normalization

  // ---- per-row sum of squares (coalesced over t) ----
  const int r = tid & 63;
  const int h = tid >> 6;
  float s = 0.f;
  #pragma unroll 8
  for (int d = h * 64; d < h * 64 + 64; ++d) {
    float v = xbase[(size_t)d * T_DIM + r];
    s += v * v;
  }
  rowpart[tid] = s;
  __syncthreads();
  float scale = 1.0f / fmaxf(sqrtf(rowpart[r] + rowpart[r + 64]), 1e-12f);
  #pragma unroll 8
  for (int d = h * 64; d < h * 64 + 64; ++d) {
    xh[r * ROW_STRIDE + d] = (_Float16)(xbase[(size_t)d * T_DIM + r] * scale);
  }
  wait_async0();
  __syncthreads();

  const int lane   = tid & 31;
  const int wave   = tid >> 5;
  const int lrow   = lane & 15;
  const int hiHalf = lane >> 4;
  const int myrow  = wave * 16 + lrow;

  // A fragments (16x32 f16 ISA layout), register-resident for the whole K sweep
  v16h a[4];
  #pragma unroll
  for (int c = 0; c < 4; ++c) {
    const _Float16* p = &xh[myrow * ROW_STRIDE + c * 32 + hiHalf * 8];
    v8h lo = *(const v8h*)p;
    v8h hi = *(const v8h*)(p + 16);
    a[c] = __builtin_shufflevector(lo, hi, 0,1,2,3,4,5,6,7,8,9,10,11,12,13,14,15);
  }

  float bestv[8];
  int   besti[8];
  #pragma unroll
  for (int q = 0; q < 8; ++q) { bestv[q] = -3.0e38f; besti[q] = 0; }

  for (int chunk = 0; chunk < NCHUNK; ++chunk) {
    const int cur = chunk & 1;
    if (chunk + 1 < NCHUNK) stage(chunk + 1, cur ^ 1);   // prefetch next chunk

    const _Float16* whc = wh + cur * (CODES_PER_CHUNK * ROW_STRIDE);
    v8f acc[4];
    #pragma unroll
    for (int tile = 0; tile < 4; ++tile) acc[tile] = v8f{};

    #pragma unroll
    for (int c = 0; c < 4; ++c) {
      v16h bt[4];                        // group all B loads ahead of the WMMAs
      #pragma unroll
      for (int tile = 0; tile < 4; ++tile) {
        const _Float16* p = &whc[(tile * 16 + lrow) * ROW_STRIDE + c * 32 + hiHalf * 8];
        v8h lo = *(const v8h*)p;
        v8h hi = *(const v8h*)(p + 16);
        bt[tile] = __builtin_shufflevector(lo, hi, 0,1,2,3,4,5,6,7,8,9,10,11,12,13,14,15);
      }
      #pragma unroll
      for (int tile = 0; tile < 4; ++tile) {
        acc[tile] = __builtin_amdgcn_wmma_f32_16x16x32_f16(
            false, a[c], false, bt[tile], (short)0, acc[tile], false, false);
      }
    }

    #pragma unroll
    for (int tile = 0; tile < 4; ++tile) {
      const int col = chunk * CODES_PER_CHUNK + tile * 16 + lrow;
      #pragma unroll
      for (int q = 0; q < 8; ++q) {
        if (acc[tile][q] > bestv[q]) { bestv[q] = acc[tile][q]; besti[q] = col; }
      }
    }

    wait_async0();                       // prefetch done before next iteration reads it
    __syncthreads();                     // all waves done reading `cur`
  }

  // reduce over the 16 lanes holding each row (xor masks stay within halves)
  #pragma unroll
  for (int m = 1; m <= 8; m <<= 1) {
    #pragma unroll
    for (int q = 0; q < 8; ++q) {
      float ov = __shfl_xor(bestv[q], m, 32);
      int   oi = __shfl_xor(besti[q], m, 32);
      if (ov > bestv[q] || (ov == bestv[q] && oi < besti[q])) {
        bestv[q] = ov; besti[q] = oi;
      }
    }
  }
  if (lrow == 0) {                       // lane 0: rows 0-7, lane 16: rows 8-15
    int base = n0 + wave * 16 + hiHalf * 8;
    #pragma unroll
    for (int q = 0; q < 8; ++q) out_idx[base + q] = besti[q];
  }
}

// ---------------------------------------------------------------------------
// Kernel 3: gather quantized values in [B,D,T] order (x4 vectorized) + loss.
// ---------------------------------------------------------------------------
__global__ __launch_bounds__(256) void k_gather_loss(
    const float* __restrict__ x, const float* __restrict__ W,
    const int* __restrict__ idx, float* __restrict__ out_q,
    float* __restrict__ out_idx_f, float* __restrict__ loss_accum) {
  __shared__ float red[256];
  const int tid = threadIdx.x;
  const size_t e = ((size_t)blockIdx.x * 256 + tid) * 4;   // flat [b][d][t], t fastest
  const int t  = (int)(e & (T_DIM - 1));
  const int bd = (int)(e >> 11);
  const int d  = bd & (D_DIM - 1);
  const int b  = bd >> 7;
  const int n  = b * T_DIM + t;

  float4 xv = *(const float4*)(x + e);
  int k0 = idx[n + 0], k1 = idx[n + 1], k2 = idx[n + 2], k3 = idx[n + 3];
  float4 q;
  q.x = W[(size_t)k0 * D_DIM + d];
  q.y = W[(size_t)k1 * D_DIM + d];
  q.z = W[(size_t)k2 * D_DIM + d];
  q.w = W[(size_t)k3 * D_DIM + d];
  *(float4*)(out_q + e) = q;             // straight-through forward value == quantized
  if (d == 0) {
    out_idx_f[n + 0] = (float)k0;
    out_idx_f[n + 1] = (float)k1;
    out_idx_f[n + 2] = (float)k2;
    out_idx_f[n + 3] = (float)k3;
  }
  float d0 = q.x - xv.x, d1 = q.y - xv.y, d2 = q.z - xv.z, d3 = q.w - xv.w;
  red[tid] = d0 * d0 + d1 * d1 + d2 * d2 + d3 * d3;
  __syncthreads();
  #pragma unroll
  for (int s = 128; s > 0; s >>= 1) {
    if (tid < s) red[tid] += red[tid + s];
    __syncthreads();
  }
  if (tid == 0) atomicAdd(loss_accum, red[0]);
}

// ---------------------------------------------------------------------------
// Kernel 4: loss = (1 + 0.1) * mean((q - x)^2)
// ---------------------------------------------------------------------------
__global__ void k_finalize(const float* __restrict__ loss_accum, float* __restrict__ out_loss) {
  out_loss[0] = loss_accum[0] * (1.1f / (float)(B_DIM * T_DIM * D_DIM));
}

extern "C" void kernel_launch(void* const* d_in, const int* in_sizes, int n_in,
                              void* d_out, int out_size, void* d_ws, size_t ws_size,
                              hipStream_t stream) {
  const float* x = (const float*)d_in[0];          // [B, D, T] f32
  const float* W = (const float*)d_in[1];          // [K, D] f32

  float* out      = (float*)d_out;
  float* out_loss = out;
  float* out_q    = out + 1;
  float* out_idx  = out + 1 + (size_t)B_DIM * D_DIM * T_DIM;

  _Float16* wn   = (_Float16*)d_ws;
  int*      idxb = (int*)((char*)d_ws + (size_t)K_DIM * D_DIM * sizeof(_Float16));
  float*    lacc = (float*)((char*)d_ws + (size_t)K_DIM * D_DIM * sizeof(_Float16)
                                        + (size_t)N_DIM * sizeof(int));

  k_norm_codebook<<<K_DIM, 128, 0, stream>>>(W, wn, lacc);
  k_vq_argmax<<<N_DIM / ROWS_PER_BLOCK, THREADS, 0, stream>>>(x, wn, idxb);
  k_gather_loss<<<(B_DIM * D_DIM * T_DIM) / 1024, 256, 0, stream>>>(
      x, W, idxb, out_q, out_idx, lacc);
  k_finalize<<<1, 1, 0, stream>>>(lacc, out_loss);
}
